// AMKPDModel_58128087384435
// MI455X (gfx1250) — matrix operand order repaired and checked by reading, not compiled, not run on hardware
//
#include <hip/hip_runtime.h>
#include <hip/hip_bf16.h>
#include <cstdint>

// MI455X (gfx1250) implementation.
// - All dense GEMMs use v_wmma_f32_16x16x32_f16 (wave32 WMMA).
// - Double-buffered LDS staging via global_load_async_to_lds_b128 + s_wait_asynccnt
//   (guarded by __has_builtin; falls back to sync copies). A staged as raw f32,
//   converted with v_cvt_pk_f16_f32 during fragment build so cvt co-issues with WMMA.
// - Spectral norm simplified: sigma = ||W v|| (u2 = Wv/||Wv|| => u2.(Wv) = ||Wv||).
//   hqW/hkW (536MB) handled by coalesced bandwidth-bound matvec kernels.

typedef __attribute__((ext_vector_type(16))) _Float16 v16h;
typedef __attribute__((ext_vector_type(8)))  _Float16 v8h;
typedef __attribute__((ext_vector_type(8)))  float    v8f;

#if defined(__gfx1250__) && __has_builtin(__builtin_amdgcn_global_load_async_to_lds_b128) && __has_builtin(__builtin_amdgcn_s_wait_asynccnt)
#define HAVE_ASYNC_LDS 1
#else
#define HAVE_ASYNC_LDS 0
#endif

namespace {

constexpr int   Vv     = 32000;
constexpr int   DM     = 512;
constexpr int   DS     = 64;
constexpr int   KBLK   = 4;
constexpr int   INNER  = 2048;
constexpr int   Bb     = 4;
constexpr int   Nn     = 512;
constexpr int   LOOPS  = 6;
constexpr int   ROWS   = Bb * Nn;        // 2048
constexpr long  HQROWS = (long)DM * DS;  // 32768

enum { EPI_PLAIN = 0, EPI_BIAS_ELU1, EPI_RELU_SQ_ROWSUM, EPI_QI, EPI_ADD_SRC, EPI_SILU, EPI_MULC };
enum { BL_NT = 0, BL_NN = 1 };

struct EpiArgs {
  const float* bias;     // per-column bias
  const float* src;      // residual source (same shape as C)
  float*       rowsum;   // per-row accumulator (Nrm)
  const float* dtptr;    // dt scalar for softplus
  long         rowsumBS; // batch stride for rowsum
};

struct alignas(8) H4 { _Float16 h[4]; };

// Pointee type the async-LDS builtin expects: 16B int vector.
typedef int v4i_a __attribute__((vector_size(16)));

// 16B global -> LDS copy; async on gfx1250, plain VGPR copy otherwise.
__device__ __forceinline__ void copy16_g2l(void* l, const void* g) {
#if HAVE_ASYNC_LDS
  __builtin_amdgcn_global_load_async_to_lds_b128((v4i_a*)g, (v4i_a*)l, 0, 0);
#else
  *(float4*)l = *(const float4*)g;
#endif
}
template <int Ncnt>
__device__ __forceinline__ void wait_async() {
#if HAVE_ASYNC_LDS
  __builtin_amdgcn_s_wait_asynccnt((unsigned short)Ncnt);
#endif
}

// ---------------------------------------------------------------------------
// Tiled WMMA GEMM:  C[M,N] = A[M,K] * op(B) with fused epilogue.
// BL_NT: B is [N,K] row-major (X @ W.T).  BL_NN: B is [K,N] row-major.
// Block tile 128x64, 8 waves, each wave 32x32 (2x2 wmma tiles), K-step 32,
// double-buffered async LDS staging.
// ---------------------------------------------------------------------------
template <int EPI, int BL, typename TB>
__global__ __launch_bounds__(256) void gemm_kernel(
    const float* __restrict__ A, const TB* __restrict__ Bm, float* __restrict__ C,
    int M, int N, int Kd, int lda, int ldb, int ldc,
    long aBS, long bBS, long cBS, EpiArgs ep) {
  constexpr int BM = 128, BN = 64, TK = 32;
  constexpr int LDA = TK + 4;  // f32 rows: stride 36 dwords -> conflict-free
  constexpr int LDB = TK + 8;  // f16 rows: stride 80B
  constexpr bool BASYNC = (BL == BL_NT) && (sizeof(TB) == 2);  // raw-copy B path
  constexpr int INFLIGHT = 4 + (BASYNC ? 1 : 0);  // async instrs per wave per stage

  __shared__ float    Asf[2][BM][LDA];
  __shared__ _Float16 Bs[2][BN][LDB];

  const int tid  = threadIdx.x;
  const int bz   = blockIdx.z;
  const float* Ag = A + (long)bz * aBS;
  const TB*    Bg = Bm + (long)bz * bBS;
  float*       Cg = C + (long)bz * cBS;
  const int row0 = blockIdx.y * BM;
  const int col0 = blockIdx.x * BN;
  const int w = tid >> 5, lane = tid & 31;
  const int waveM = (w >> 1) * 32, waveN = (w & 1) * 32;

  auto stageA = [&](int buf, int kb) {
    // 128 rows x 128B : 1024 x 16B chunks, 4 per thread
#pragma unroll
    for (int s = 0; s < 4; s++) {
      const int idx = tid + s * 256;
      const int r = idx >> 3, ch = idx & 7;
      copy16_g2l(&Asf[buf][r][ch * 4], Ag + (long)(row0 + r) * lda + kb + ch * 4);
    }
  };
  auto stageB = [&](int buf, int kb) {
    if constexpr (BASYNC) {
      // 64 rows x 64B : 256 x 16B chunks, 1 per thread
      const int r = tid >> 2, ch = tid & 3;
      copy16_g2l(&Bs[buf][r][ch * 8], Bg + (long)(col0 + r) * ldb + kb + ch * 8);
    } else if constexpr (BL == BL_NT) {  // f32 weights/activations: convert
      const int r = tid >> 3, c4 = (tid & 7) * 4;
#pragma unroll
      for (int rr = 0; rr < BN; rr += 32) {
        float4 v = *(const float4*)(Bg + (long)(col0 + r + rr) * ldb + kb + c4);
        H4 t; t.h[0] = (_Float16)v.x; t.h[1] = (_Float16)v.y;
        t.h[2] = (_Float16)v.z; t.h[3] = (_Float16)v.w;
        *(H4*)&Bs[buf][r + rr][c4] = t;
      }
    } else {  // BL_NN: B[k][j] (f32)
      const int jj = (tid & 15) * 4, k0 = tid >> 4;
#pragma unroll
      for (int kk = 0; kk < TK; kk += 16) {
        float4 v = *(const float4*)(Bg + (long)(kb + k0 + kk) * ldb + col0 + jj);
        Bs[buf][jj + 0][k0 + kk] = (_Float16)v.x;
        Bs[buf][jj + 1][k0 + kk] = (_Float16)v.y;
        Bs[buf][jj + 2][k0 + kk] = (_Float16)v.z;
        Bs[buf][jj + 3][k0 + kk] = (_Float16)v.w;
      }
    }
  };

  v8f acc[2][2] = {};
  const int nk = Kd / TK;
  stageA(0, 0);
  stageB(0, 0);
  for (int i = 0; i < nk; i++) {
    const int buf = i & 1;
    if (i + 1 < nk) {
      stageA(buf ^ 1, (i + 1) * TK);
      stageB(buf ^ 1, (i + 1) * TK);
      wait_async<INFLIGHT>();  // in-order: stage i has landed
    } else {
      wait_async<0>();
    }
    __syncthreads();

    // ---- fragments per documented CDNA5 layouts ----
    // A 16x32: lane<16 -> K0..7 / K16..23 ; lane>=16 -> K8..15 / K24..31
    const int mh = lane & 15;
    const int kh = (lane >> 4) * 8;
    v16h af[2], bf[2];
#pragma unroll
    for (int t = 0; t < 2; t++) {
      const float* ar = &Asf[buf][waveM + t * 16 + mh][0];
      float4 q0 = *(const float4*)(ar + kh);
      float4 q1 = *(const float4*)(ar + kh + 4);
      float4 q2 = *(const float4*)(ar + 16 + kh);
      float4 q3 = *(const float4*)(ar + 20 + kh);
      v16h a;
      a[0] = (_Float16)q0.x;  a[1] = (_Float16)q0.y;
      a[2] = (_Float16)q0.z;  a[3] = (_Float16)q0.w;
      a[4] = (_Float16)q1.x;  a[5] = (_Float16)q1.y;
      a[6] = (_Float16)q1.z;  a[7] = (_Float16)q1.w;
      a[8] = (_Float16)q2.x;  a[9] = (_Float16)q2.y;
      a[10] = (_Float16)q2.z; a[11] = (_Float16)q2.w;
      a[12] = (_Float16)q3.x; a[13] = (_Float16)q3.y;
      a[14] = (_Float16)q3.z; a[15] = (_Float16)q3.w;
      af[t] = a;
      // B 32x16: lane = N (mod 16); lanes 0-15 K0..7 / K16..23, lanes 16-31 K8..15 / K24..31
      const _Float16* br = &Bs[buf][waveN + t * 16 + mh][0];
      const int kbase = (lane >> 4) * 8;
      v8h bl = *(const v8h*)(br + kbase);
      v8h bh = *(const v8h*)(br + kbase + 16);
      bf[t] = __builtin_shufflevector(bl, bh, 0, 1, 2, 3, 4, 5, 6, 7,
                                      8, 9, 10, 11, 12, 13, 14, 15);
    }
#pragma unroll
    for (int ii = 0; ii < 2; ii++)
#pragma unroll
      for (int jj = 0; jj < 2; jj++)
        acc[ii][jj] = __builtin_amdgcn_wmma_f32_16x16x32_f16(
            false, af[ii], false, bf[jj], (short)0, acc[ii][jj], false, false);
    __syncthreads();
  }

  // ---- epilogue + store (C layout: lane = n&15, vgpr r -> m = r + 8*(lane>>4)) ----
  const int ln15 = lane & 15, mo = (lane >> 4) * 8;
  float sp = 0.f;
  if constexpr (EPI == EPI_QI) sp = log1pf(__expf(ep.dtptr[0]));
#pragma unroll
  for (int ti = 0; ti < 2; ti++) {
#pragma unroll
    for (int tj = 0; tj < 2; tj++) {
      const int ccol = col0 + waveN + tj * 16 + ln15;
#pragma unroll
      for (int r = 0; r < 8; r++) {
        const int crow = row0 + waveM + ti * 16 + mo + r;
        const long cidx = (long)crow * ldc + ccol;
        float x = acc[ti][tj][r];
        float o;
        if constexpr (EPI == EPI_PLAIN) {
          o = x;
        } else if constexpr (EPI == EPI_BIAS_ELU1) {
          float z = x + ep.bias[ccol];
          o = (z > 0.f) ? z + 1.f : __expf(z);  // elu(z)+1
        } else if constexpr (EPI == EPI_RELU_SQ_ROWSUM) {
          float t = x > 0.f ? x : 0.f;
          o = t * t;
          float s = o;
          s += __shfl_xor(s, 1, 32); s += __shfl_xor(s, 2, 32);
          s += __shfl_xor(s, 4, 32); s += __shfl_xor(s, 8, 32);
          if (ln15 == 0) atomicAdd(ep.rowsum + (long)bz * ep.rowsumBS + crow, s);
        } else if constexpr (EPI == EPI_QI) {
          o = ep.src[cidx] + sp * (x + ep.bias[ccol]);
        } else if constexpr (EPI == EPI_ADD_SRC) {
          o = ep.src[cidx] + x;
        } else if constexpr (EPI == EPI_SILU) {
          o = x / (1.f + __expf(-x));
        } else {  // EPI_MULC
          o = Cg[cidx] * x;
        }
        Cg[cidx] = o;
      }
    }
  }
}

// ---------------------------------------------------------------------------
// Reductions / LN / elementwise
// ---------------------------------------------------------------------------
__device__ __forceinline__ float block_reduce_256(float v, float* red) {
  const int lane = threadIdx.x & 31, w = threadIdx.x >> 5;
#pragma unroll
  for (int off = 16; off > 0; off >>= 1) v += __shfl_down(v, off, 32);
  __syncthreads();
  if (lane == 0) red[w] = v;
  __syncthreads();
  if (w == 0) {
    float r = (lane < 8) ? red[lane] : 0.f;
#pragma unroll
    for (int off = 4; off > 0; off >>= 1) r += __shfl_down(r, off, 32);
    if (lane == 0) red[0] = r;
  }
  __syncthreads();
  return red[0];
}

template <bool ACCQP>
__global__ __launch_bounds__(256) void ln_kernel(
    const float* __restrict__ in, const float* __restrict__ g,
    const float* __restrict__ b, float* __restrict__ out, float* __restrict__ qp) {
  __shared__ float red[8];
  const int row = blockIdx.x, t = threadIdx.x;
  const float* ip = in + (long)row * DM;
  float x0 = ip[t], x1 = ip[t + 256];
  float mu = block_reduce_256(x0 + x1, red) * (1.f / DM);
  float d0 = x0 - mu, d1 = x1 - mu;
  float var = block_reduce_256(d0 * d0 + d1 * d1, red) * (1.f / DM);
  float rs = rsqrtf(var + 1e-5f);
  float y0 = d0 * rs * g[t] + b[t];
  float y1 = d1 * rs * g[t + 256] + b[t + 256];
  float* op = out + (long)row * DM;
  op[t] = y0; op[t + 256] = y1;
  if constexpr (ACCQP) {
    const int bb = row >> 9;
    atomicAdd(qp + bb * DM + t, y0 * (1.f / Nn));
    atomicAdd(qp + bb * DM + t + 256, y1 * (1.f / Nn));
  }
}

__global__ __launch_bounds__(256) void prep_kernel(
    const int* __restrict__ inputs, const unsigned char* __restrict__ halted,
    const int* __restrict__ carry_inputs, const float* __restrict__ carry_hidden,
    const float* __restrict__ embedding, const float* __restrict__ pos_emb,
    const float* __restrict__ in_g, const float* __restrict__ in_b,
    const float* __restrict__ init_hidden,
    float* __restrict__ X, float* __restrict__ Qcur) {
  __shared__ float red[8];
  const int row = blockIdx.x, t = threadIdx.x;
  const int b = row >> 9, n = row & (Nn - 1);
  const int h = halted[b] ? 1 : 0;
  const int tok = h ? inputs[row] : carry_inputs[row];
  const float* er = embedding + (long)tok * DM;
  const float* pr = pos_emb + (long)n * DM;
  float x0 = er[t] + pr[t], x1 = er[t + 256] + pr[t + 256];
  float mu = block_reduce_256(x0 + x1, red) * (1.f / DM);
  float d0 = x0 - mu, d1 = x1 - mu;
  float var = block_reduce_256(d0 * d0 + d1 * d1, red) * (1.f / DM);
  float rs = rsqrtf(var + 1e-5f);
  float y0 = d0 * rs * in_g[t] + in_b[t];
  float y1 = d1 * rs * in_g[t + 256] + in_b[t + 256];
  X[(long)row * DM + t] = y0;
  X[(long)row * DM + t + 256] = y1;
  float h0 = h ? init_hidden[t] : carry_hidden[(long)row * DM + t];
  float h1 = h ? init_hidden[t + 256] : carry_hidden[(long)row * DM + t + 256];
  Qcur[(long)row * DM + t] = h0 + y0;
  Qcur[(long)row * DM + t + 256] = h1 + y1;
}

// v_partial = W.T @ u  (W: 32768x512 row-major). One WG covers 64 rows; thread
// t owns columns t and t+256 -> zero-conflict register accumulation + atomics.
__global__ __launch_bounds__(256) void colreduce_kernel(
    const float* __restrict__ W, const float* __restrict__ u, float* __restrict__ v) {
  const int t = threadIdx.x;
  const long base = (long)blockIdx.x * 64;
  float a0 = 0.f, a1 = 0.f;
  for (int r = 0; r < 64; r++) {
    const float uu = u[base + r];
    const float* wr = W + (base + r) * DM;
    a0 += wr[t] * uu;
    a1 += wr[t + 256] * uu;
  }
  atomicAdd(v + t, a0);
  atomicAdd(v + t + 256, a1);
}

__global__ __launch_bounds__(256) void vnorm_kernel(float* __restrict__ v) {
  __shared__ float red[8];
  const int t = threadIdx.x;
  float a = v[t], b = v[t + 256];
  float ss = block_reduce_256(a * a + b * b, red);
  float inv = 1.f / (sqrtf(ss) + 1e-12f);
  v[t] = a * inv;
  v[t + 256] = b * inv;
}

// sigma^2 = sum_i (W[i,:].v)^2  -> sigma = ||W v||  (one wave per row)
__global__ __launch_bounds__(256) void sigma_kernel(
    const float* __restrict__ W, const float* __restrict__ v, float* __restrict__ sigsq) {
  const int lane = threadIdx.x & 31;
  const long row = (long)blockIdx.x * 8 + (threadIdx.x >> 5);
  const float* wr = W + row * DM;
  float acc = 0.f;
#pragma unroll
  for (int c0 = 0; c0 < DM; c0 += 128) {
    float4 wv = *(const float4*)(wr + c0 + lane * 4);
    float4 vv = *(const float4*)(v + c0 + lane * 4);
    acc += wv.x * vv.x + wv.y * vv.y + wv.z * vv.z + wv.w * vv.w;
  }
#pragma unroll
  for (int off = 16; off > 0; off >>= 1) acc += __shfl_down(acc, off, 32);
  if (lane == 0) atomicAdd(sigsq, acc * acc);
}

// O[b, row] = (W[row,:] . qp[b,:]) / sigma    (all 4 batches in one W pass)
__global__ __launch_bounds__(256) void oqproj_kernel(
    const float* __restrict__ W, const float* __restrict__ qp,
    const float* __restrict__ sigsq, float* __restrict__ O) {
  const int lane = threadIdx.x & 31;
  const long row = (long)blockIdx.x * 8 + (threadIdx.x >> 5);
  const float* wr = W + row * DM;
  float acc[Bb] = {};
#pragma unroll
  for (int c0 = 0; c0 < DM; c0 += 128) {
    float4 wv = *(const float4*)(wr + c0 + lane * 4);
#pragma unroll
    for (int b = 0; b < Bb; b++) {
      float4 qv = *(const float4*)(qp + b * DM + c0 + lane * 4);
      acc[b] += wv.x * qv.x + wv.y * qv.y + wv.z * qv.z + wv.w * qv.w;
    }
  }
#pragma unroll
  for (int b = 0; b < Bb; b++) {
    float a = acc[b];
#pragma unroll
    for (int off = 16; off > 0; off >>= 1) a += __shfl_down(a, off, 32);
    acc[b] = a;
  }
  if (lane == 0) {
    const float rsig = rsqrtf(fmaxf(sigsq[0], 1e-24f));
#pragma unroll
    for (int b = 0; b < Bb; b++) O[(long)b * HQROWS + row] = acc[b] * rsig;
  }
}

__global__ void fill_kernel(float* __restrict__ p, float v, long n) {
  long i = (long)blockIdx.x * 256 + threadIdx.x;
  if (i < n) p[i] = v;
}
__global__ void cvt_f16_kernel(const float* __restrict__ s, _Float16* __restrict__ d, long n) {
  long i = (long)blockIdx.x * 256 + threadIdx.x;
  if (i < n) d[i] = (_Float16)s[i];
}
__global__ void add_kernel(float* __restrict__ d, const float* __restrict__ s, long n) {
  long i = (long)blockIdx.x * 256 + threadIdx.x;
  if (i < n) d[i] += s[i];
}
// m = Att / Nrm[row] - Qn   (in place over Att)
__global__ void m_kernel(float* __restrict__ att, const float* __restrict__ nrm,
                         const float* __restrict__ qn, long n) {
  long i = (long)blockIdx.x * 256 + threadIdx.x;
  if (i < n) {
    const long row = i >> 9;  // /DM
    att[i] = att[i] / nrm[row] - qn[i];
  }
}
// depthwise conv, kernel 3, pad 1, over sequence dim
__global__ void dwconv_kernel(const float* __restrict__ H, const float* __restrict__ w3,
                              float* __restrict__ out) {
  long i = (long)blockIdx.x * 256 + threadIdx.x;
  if (i >= (long)ROWS * INNER) return;
  const int c = (int)(i & (INNER - 1));
  const long rown = i >> 11;  // /INNER -> b*N+n
  const int n = (int)(rown & (Nn - 1));
  const float w0 = w3[c * 3 + 0], w1 = w3[c * 3 + 1], w2 = w3[c * 3 + 2];
  float s = H[i] * w1;
  if (n > 0) s += H[i - INNER] * w0;
  if (n < Nn - 1) s += H[i + INNER] * w2;
  out[i] = s;
}
__global__ void qfmean_kernel(const float* __restrict__ Qf, float* __restrict__ qm) {
  const int i = blockIdx.x * 256 + threadIdx.x;  // 0..2047
  const int b = i >> 9, d = i & (DM - 1);
  const float* p = Qf + ((long)b * Nn) * DM + d;
  float s = 0.f;
  for (int n = 0; n < Nn; n++) s += p[(long)n * DM];
  qm[i] = s * (1.f / Nn);
}
__global__ __launch_bounds__(256) void halt_kernel(
    const float* __restrict__ qm, const float* __restrict__ hW, const float* __restrict__ hb,
    const unsigned char* __restrict__ halted_in, const int* __restrict__ steps_in,
    float* __restrict__ out_halt, float* __restrict__ out_steps, float* __restrict__ out_halted) {
  __shared__ float red[8];
  const int b = blockIdx.x, t = threadIdx.x;
  float s = qm[b * DM + t] * hW[t] + qm[b * DM + t + 256] * hW[t + 256];
  s = block_reduce_256(s, red);
  if (t == 0) {
    out_halt[b] = s + hb[0];
    const int st = (halted_in[b] ? 0 : steps_in[b]) + 1;
    out_steps[b] = (float)st;
    out_halted[b] = (st >= LOOPS) ? 1.f : 0.f;
  }
}

// ---------------------------------------------------------------------------
// Host helpers
// ---------------------------------------------------------------------------
inline long cdiv(long a, long b) { return (a + b - 1) / b; }

template <int EPI, int BL, typename TB>
void gemm_go(const float* A, const TB* B, float* C, int M, int N, int K,
             int lda, int ldb, int ldc, long aBS, long bBS, long cBS,
             EpiArgs ep, int batch, hipStream_t s) {
  dim3 grid(N / 64, M / 128, batch);
  gemm_kernel<EPI, BL, TB><<<grid, 256, 0, s>>>(A, B, C, M, N, K, lda, ldb, ldc,
                                                aBS, bBS, cBS, ep);
}

}  // namespace

extern "C" void kernel_launch(void* const* d_in, const int* in_sizes, int n_in,
                              void* d_out, int out_size, void* d_ws, size_t ws_size,
                              hipStream_t stream) {
  (void)in_sizes; (void)n_in; (void)out_size; (void)ws_size;
  // ---- inputs (setup_inputs order) ----
  const int*   inputs       = (const int*)d_in[0];
  const unsigned char* halted = (const unsigned char*)d_in[2];
  const int*   steps        = (const int*)d_in[3];
  const float* carry_hidden = (const float*)d_in[4];
  const int*   carry_inputs = (const int*)d_in[5];
  const float* embedding    = (const float*)d_in[7];
  const float* pos_emb      = (const float*)d_in[8];
  const float* in_g         = (const float*)d_in[9];
  const float* in_b         = (const float*)d_in[10];
  const float* init_hidden  = (const float*)d_in[11];
  const float* dt           = (const float*)d_in[12];
  const float* mW           = (const float*)d_in[13];
  const float* mb           = (const float*)d_in[14];
  const float* hqW          = (const float*)d_in[15];
  const float* hkW          = (const float*)d_in[16];
  const float* uq           = (const float*)d_in[17];
  const float* uk           = (const float*)d_in[18];
  const float* BQ           = (const float*)d_in[19];
  const float* BK           = (const float*)d_in[20];
  const float* g1           = (const float*)d_in[21];
  const float* b1           = (const float*)d_in[22];
  const float* g2           = (const float*)d_in[23];
  const float* b2           = (const float*)d_in[24];
  const float* WupW         = (const float*)d_in[25];
  const float* convW        = (const float*)d_in[26];
  const float* WdownW       = (const float*)d_in[27];
  const float* fin_g        = (const float*)d_in[28];
  const float* fin_b        = (const float*)d_in[29];
  const float* halt_W       = (const float*)d_in[30];
  const float* halt_b       = (const float*)d_in[31];
  const float* lm_head      = (const float*)d_in[32];
  float* out = (float*)d_out;

  // ---- output layout (floats) ----
  const long OFF_HALT   = (long)ROWS * Vv;
  const long OFF_Q      = OFF_HALT + Bb;
  const long OFF_STEPS  = OFF_Q + (long)ROWS * DM;
  const long OFF_HALTED = OFF_STEPS + Bb;

  // ---- workspace bump allocator ----
  char* wp = (char*)d_ws;
  auto alloc = [&](size_t bytes) -> void* {
    void* p = wp;
    wp += (bytes + 255) & ~(size_t)255;
    return p;
  };
  float* X    = (float*)alloc((long)ROWS * DM * 4);
  float* Qcur = (float*)alloc((long)ROWS * DM * 4);
  float* Qn   = (float*)alloc((long)ROWS * DM * 4);   // also Qn2 / Qf
  float* Qi   = (float*)alloc((long)ROWS * DM * 4);
  float* AttM = (float*)alloc((long)ROWS * DM * 4);   // Att, then m (in place)
  float* Wm   = (float*)alloc((long)Bb * Nn * Nn * 4);
  float* Hbuf = (float*)alloc((long)ROWS * INNER * 4);
  float* Hc   = (float*)alloc((long)ROWS * INNER * 4);
  float* Pq   = (float*)alloc((long)Bb * Nn * DS * 4);
  float* Pk   = (float*)alloc((long)Bb * Nn * DS * 4);
  float* Oq   = (float*)alloc((long)Bb * HQROWS * 4);
  float* Ok   = (float*)alloc((long)Bb * HQROWS * 4);
  float* qp   = (float*)alloc((long)Bb * DM * 4);
  float* Nrm  = (float*)alloc((long)Bb * Nn * 4);
  float* vtmp = (float*)alloc(DM * 4);
  float* sigs = (float*)alloc(8 * 4);                 // sigma^2: q[0..3], k[4..7]
  float* qfm  = (float*)alloc((long)Bb * DM * 4);
  _Float16* mW16  = (_Float16*)alloc((long)KBLK * DM * DM * 2);
  _Float16* Wup16 = (_Float16*)alloc((long)KBLK * 2 * INNER * DM * 2);
  _Float16* Wdn16 = (_Float16*)alloc((long)KBLK * DM * INNER * 2);
  _Float16* lm16  = (_Float16*)alloc((long)Vv * DM * 2);

  // ---- stage static weights to f16 (halves GEMM B-traffic) ----
  {
    long n;
    n = (long)KBLK * DM * DM;
    cvt_f16_kernel<<<cdiv(n, 256), 256, 0, stream>>>(mW, mW16, n);
    n = (long)KBLK * 2 * INNER * DM;
    cvt_f16_kernel<<<cdiv(n, 256), 256, 0, stream>>>(WupW, Wup16, n);
    n = (long)KBLK * DM * INNER;
    cvt_f16_kernel<<<cdiv(n, 256), 256, 0, stream>>>(WdownW, Wdn16, n);
    n = (long)Vv * DM;
    cvt_f16_kernel<<<cdiv(n, 256), 256, 0, stream>>>(lm_head, lm16, n);
  }

  // ---- spectral-norm sigmas (Q-independent: once per call) ----
  for (int k = 0; k < KBLK; k++) {
    for (int qk = 0; qk < 2; qk++) {
      const float* W = (qk == 0 ? hqW : hkW) + (long)k * HQROWS * DM;
      const float* U = (qk == 0 ? uq : uk) + (long)k * HQROWS;
      float* sg = sigs + qk * KBLK + k;
      fill_kernel<<<cdiv(DM, 256), 256, 0, stream>>>(vtmp, 0.f, DM);
      colreduce_kernel<<<HQROWS / 64, 256, 0, stream>>>(W, U, vtmp);
      vnorm_kernel<<<1, 256, 0, stream>>>(vtmp);
      fill_kernel<<<1, 256, 0, stream>>>(sg, 0.f, 1);
      sigma_kernel<<<HQROWS / 8, 256, 0, stream>>>(W, vtmp, sg);
    }
  }

  // ---- embed + LN + carry select ----
  prep_kernel<<<ROWS, 256, 0, stream>>>(inputs, halted, carry_inputs, carry_hidden,
                                        embedding, pos_emb, in_g, in_b, init_hidden,
                                        X, Qcur);

  const EpiArgs e0{};
  const long QSZ = (long)ROWS * DM;

  for (int h = 0; h < 2; h++) {  // H_CYCLES
    if (h > 0) add_kernel<<<cdiv(QSZ, 256), 256, 0, stream>>>(Qcur, X, QSZ);
    for (int k = 0; k < KBLK; k++) {
      const long wOff = (long)k * HQROWS * DM;
      // LN1 + qp (mean over sequence)
      fill_kernel<<<cdiv((long)Bb * DM, 256), 256, 0, stream>>>(qp, 0.f, (long)Bb * DM);
      ln_kernel<true><<<ROWS, 256, 0, stream>>>(Qcur, g1 + k * DM, b1 + k * DM, Qn, qp);
      // Oq/Ok = (W @ qp_b) / sigma   (single W streaming pass each)
      oqproj_kernel<<<HQROWS / 8, 256, 0, stream>>>(hqW + wOff, qp, sigs + k, Oq);
      oqproj_kernel<<<HQROWS / 8, 256, 0, stream>>>(hkW + wOff, qp, sigs + 4 + k, Ok);
      // Pq = elu(Qn @ Oq + BQ) + 1 ; Pk likewise     [batched, WMMA]
      EpiArgs eq{}; eq.bias = BQ + k * DS;
      gemm_go<EPI_BIAS_ELU1, BL_NN, float>(Qn, Oq, Pq, Nn, DS, DM, DM, DS, DS,
                                           (long)Nn * DM, HQROWS, (long)Nn * DS, eq, Bb, stream);
      EpiArgs ek{}; ek.bias = BK + k * DS;
      gemm_go<EPI_BIAS_ELU1, BL_NN, float>(Qn, Ok, Pk, Nn, DS, DM, DM, DS, DS,
                                           (long)Nn * DM, HQROWS, (long)Nn * DS, ek, Bb, stream);
      // Wm = relu(Pq @ Pk.T)^2, Nrm = 1 + rowsum(Wm)
      fill_kernel<<<cdiv((long)Bb * Nn, 256), 256, 0, stream>>>(Nrm, 1.f, (long)Bb * Nn);
      EpiArgs ew{}; ew.rowsum = Nrm; ew.rowsumBS = Nn;
      gemm_go<EPI_RELU_SQ_ROWSUM, BL_NT, float>(Pq, Pk, Wm, Nn, Nn, DS, DS, DS, Nn,
                                                (long)Nn * DS, (long)Nn * DS, (long)Nn * Nn,
                                                ew, Bb, stream);
      // Att = Wm @ Qn
      gemm_go<EPI_PLAIN, BL_NN, float>(Wm, Qn, AttM, Nn, DM, Nn, Nn, DM, DM,
                                       (long)Nn * Nn, (long)Nn * DM, (long)Nn * DM,
                                       e0, Bb, stream);
      // m = Att/Nrm - Qn (in place)
      m_kernel<<<cdiv(QSZ, 256), 256, 0, stream>>>(AttM, Nrm, Qn, QSZ);
      // Qi = Qcur + softplus(dt) * (m @ mW.T + mb)
      EpiArgs em{}; em.bias = mb + k * DM; em.src = Qcur; em.dtptr = dt + k;
      gemm_go<EPI_QI, BL_NT, _Float16>(AttM, mW16 + (long)k * DM * DM, Qi,
                                       ROWS, DM, DM, DM, DM, DM, 0, 0, 0, em, 1, stream);
      // LN2
      ln_kernel<false><<<ROWS, 256, 0, stream>>>(Qi, g2 + k * DM, b2 + k * DM, Qn, nullptr);
      // H = silu(Qn2 @ Wg.T) ; H *= (Qn2 @ Wu.T)
      const _Float16* wu = Wup16 + (long)k * 2 * INNER * DM;
      gemm_go<EPI_SILU, BL_NT, _Float16>(Qn, wu, Hbuf, ROWS, INNER, DM,
                                         DM, DM, INNER, 0, 0, 0, e0, 1, stream);
      gemm_go<EPI_MULC, BL_NT, _Float16>(Qn, wu + (long)INNER * DM, Hbuf, ROWS, INNER, DM,
                                         DM, DM, INNER, 0, 0, 0, e0, 1, stream);
      // depthwise conv over sequence
      dwconv_kernel<<<cdiv((long)ROWS * INNER, 256), 256, 0, stream>>>(
          Hbuf, convW + (long)k * INNER * 3, Hc);
      // Qcur = Qi + Hc @ Wdown.T
      EpiArgs ed{}; ed.src = Qi;
      gemm_go<EPI_ADD_SRC, BL_NT, _Float16>(Hc, Wdn16 + (long)k * DM * INNER, Qcur,
                                            ROWS, DM, INNER, INNER, INNER, DM,
                                            0, 0, 0, ed, 1, stream);
    }
  }

  // ---- head: Qf = LN(Q), logits = Qf @ lm_head.T ----
  ln_kernel<false><<<ROWS, 256, 0, stream>>>(Qcur, fin_g, fin_b, Qn, nullptr);
  gemm_go<EPI_PLAIN, BL_NT, _Float16>(Qn, lm16, out, ROWS, Vv, DM, DM, DM, Vv,
                                      0, 0, 0, e0, 1, stream);
  qfmean_kernel<<<(Bb * DM) / 256, 256, 0, stream>>>(Qn, qfm);
  halt_kernel<<<Bb, 256, 0, stream>>>(qfm, halt_W, halt_b, halted, steps,
                                      out + OFF_HALT, out + OFF_STEPS, out + OFF_HALTED);
  (void)hipMemcpyAsync(out + OFF_Q, Qcur, QSZ * sizeof(float),
                       hipMemcpyDeviceToDevice, stream);
}
